// LatentLongRange_17763984736925
// MI455X (gfx1250) — matrix operand order repaired and checked by reading, not compiled, not run on hardware
//
#include <hip/hip_runtime.h>
#include <hip/hip_bf16.h>
#include <math.h>

typedef __attribute__((ext_vector_type(16))) _Float16 v16h;
typedef __attribute__((ext_vector_type(8)))  float    v8f;

#define N_ATOMS 6144
#define FDIM    128
#define NGRAPH  24
#define KE_C    14.3996f
#define TILE_J  256

__device__ __forceinline__ float fast_rcp(float x) {
    return __builtin_amdgcn_rcpf(x);            // v_rcp_f32 (1 TRANS op)
}
__device__ __forceinline__ float softplusf(float x) {
    return (x > 20.0f) ? x : log1pf(expf(x));   // rare path (16 lanes/block)
}
__device__ __forceinline__ float siluf(float x) {
    return x * fast_rcp(1.0f + __expf(-x));     // v_exp + v_rcp
}

// ---------------------------------------------------------------------------
// Kernel 0: zero the accumulators (graph-capture safe; no hipMemset allowed)
// ---------------------------------------------------------------------------
__global__ void llr_init_kernel(float* out, float* qsum, float* cnt) {
    int t = threadIdx.x;
    if (t == 0) out[0] = 0.0f;
    if (t < NGRAPH) { qsum[t] = 0.0f; cnt[t] = 0.0f; }
}

// ---------------------------------------------------------------------------
// Kernel 1: fused WMMA MLP heads.
// Block = 256 threads = 8 waves; block owns 16 atoms (rows); wave w owns
// hidden columns [16w, 16w+16). A tile (16x128) converted to f16 ONCE into 4
// register fragments, reused by both heads. K=128 -> 4 x v_wmma_f32_16x16x32_f16
// per head, fully unrolled (8 static WMMAs). Epilogue folds SiLU + layer-2
// projection via 16-lane xor-shuffle reduction.
// ---------------------------------------------------------------------------
__global__ void __launch_bounds__(256)
llr_head_kernel(const float* __restrict__ h0,
                const int*   __restrict__ batch,
                const float* __restrict__ qW1, const float* __restrict__ qb1,
                const float* __restrict__ qW2,
                const float* __restrict__ vW1, const float* __restrict__ vb1,
                const float* __restrict__ vW2, const float* __restrict__ vb2,
                float* __restrict__ outQ,   // raw q, N
                float* __restrict__ outS,   // sqrt(softplus(vp0)), N
                float* __restrict__ outR,   // softplus(vp1), N
                float* __restrict__ qsum,   // per-graph q sum, G
                float* __restrict__ cnt)    // per-graph count, G
{
    __shared__ float s_q[16];
    __shared__ float s_v0[16];
    __shared__ float s_v1[16];

    const int tid  = threadIdx.x;
    const int lane = tid & 31;
    const int wave = tid >> 5;          // 0..7 : hidden-column block
    const int row0 = blockIdx.x * 16;

    if (tid < 16) { s_q[tid] = 0.0f; s_v0[tid] = 0.0f; s_v1[tid] = 0.0f; }
    __syncthreads();

    const int m  = lane & 15;           // A row within tile
    const int hb = lane >> 4;           // half-wave select
    const float* __restrict__ arow = h0 + (size_t)(row0 + m) * FDIM;
    const int col = wave * 16 + (lane & 15);   // hidden column this lane owns

    // ---- A fragments: 16x32 f16 x 4 (ISA 7.12.2 16-bit A layout), loaded once
    v16h afr[4];
    #pragma unroll
    for (int kb = 0; kb < 4; ++kb) {
        const int k0 = kb * 32;
        #pragma unroll
        for (int v = 0; v < 8; ++v) {
            int base = (v < 4) ? (2 * v + 8 * hb) : (16 + 2 * (v - 4) + 8 * hb);
            int k = k0 + base;
            afr[kb][2 * v]     = (_Float16)arow[k];
            afr[kb][2 * v + 1] = (_Float16)arow[k + 1];
        }
    }

    #pragma unroll
    for (int head = 0; head < 2; ++head) {
        const float* __restrict__ W1 = head ? vW1 : qW1;
        const float* __restrict__ b1 = head ? vb1 : qb1;

        v8f c = {0.f, 0.f, 0.f, 0.f, 0.f, 0.f, 0.f, 0.f};

        #pragma unroll
        for (int kb = 0; kb < 4; ++kb) {
            const int k0 = kb * 32;
            // ---- B fragment: 32x16 f16, lane = column, same K order as A ----
            v16h b;
            #pragma unroll
            for (int v = 0; v < 8; ++v) {
                int k = k0 + ((v < 4) ? (2 * v + 8 * hb) : (16 + 2 * (v - 4) + 8 * hb));
                b[2 * v]     = (_Float16)W1[(size_t)k * FDIM + col];
                b[2 * v + 1] = (_Float16)W1[(size_t)(k + 1) * FDIM + col];
            }
            c = __builtin_amdgcn_wmma_f32_16x16x32_f16(
                    false, afr[kb], false, b, (short)0, c, false, false);
        }

        // ---- epilogue: bias + SiLU + project by layer-2 weight, reduce cols ----
        const float bias = b1[col];
        if (head == 0) {
            const float w = qW2[col];
            #pragma unroll
            for (int r = 0; r < 8; ++r) {
                float p = siluf(c[r] + bias) * w;
                #pragma unroll
                for (int off = 1; off < 16; off <<= 1) p += __shfl_xor(p, off, 16);
                if ((lane & 15) == 0) atomicAdd(&s_q[r + 8 * hb], p);
            }
        } else {
            const float w0 = vW2[2 * col];
            const float w1 = vW2[2 * col + 1];
            #pragma unroll
            for (int r = 0; r < 8; ++r) {
                float s  = siluf(c[r] + bias);
                float p0 = s * w0;
                float p1 = s * w1;
                #pragma unroll
                for (int off = 1; off < 16; off <<= 1) {
                    p0 += __shfl_xor(p0, off, 16);
                    p1 += __shfl_xor(p1, off, 16);
                }
                if ((lane & 15) == 0) {
                    atomicAdd(&s_v0[r + 8 * hb], p0);
                    atomicAdd(&s_v1[r + 8 * hb], p1);
                }
            }
        }
    }
    __syncthreads();

    if (tid < 16) {
        const int row = row0 + tid;
        const float q = s_q[tid];
        outQ[row] = q;
        const int g = batch[row];
        atomicAdd(&qsum[g], q);
        atomicAdd(&cnt[g], 1.0f);
        const float vp0 = s_v0[tid] + vb2[0];
        const float vp1 = s_v1[tid] + vb2[1];
        outS[row] = sqrtf(softplusf(vp0));   // sqrt(c6_i): sqrt(c6_i c6_j)=s_i*s_j
        outR[row] = softplusf(vp1);          // rv_i: rv_ij^6 = (rv_i rv_j)^3
    }
}

// ---------------------------------------------------------------------------
// Kernel 2: pairwise masked energy, LDS-tiled over j, scalar atomicAdd.
// Inner loop is pure VALU: v_rcp_f32 instead of IEEE division (2x per pair).
// ---------------------------------------------------------------------------
__global__ void __launch_bounds__(256)
llr_pair_kernel(const float* __restrict__ pos,
                const int*   __restrict__ batch,
                const float* __restrict__ wQ,
                const float* __restrict__ wS,
                const float* __restrict__ wR,
                const float* __restrict__ qsum,
                const float* __restrict__ cnt,
                const float* __restrict__ sigma,
                float* __restrict__ out)
{
    __shared__ float s_mean[NGRAPH];
    __shared__ float jx[TILE_J], jy[TILE_J], jz[TILE_J];
    __shared__ float jq[TILE_J], js[TILE_J], jr[TILE_J];
    __shared__ int   jb[TILE_J];
    __shared__ float red[256];

    const int tid = threadIdx.x;
    if (tid < NGRAPH) s_mean[tid] = qsum[tid] * fast_rcp(fmaxf(cnt[tid], 1.0f));
    __syncthreads();

    const int i  = blockIdx.x * 256 + tid;
    const int bi = batch[i];
    const float xi = pos[3 * i], yi = pos[3 * i + 1], zi = pos[3 * i + 2];
    const float qi = wQ[i] - s_mean[bi];
    const float si = wS[i];
    const float ri = wR[i];
    const float inv_s = fast_rcp(1.41421356f * sigma[0]);

    float acc = 0.0f;
    for (int jt = 0; jt < N_ATOMS / TILE_J; ++jt) {
        __syncthreads();
        const int j = jt * TILE_J + tid;
        const int bj = batch[j];
        jx[tid] = pos[3 * j];
        jy[tid] = pos[3 * j + 1];
        jz[tid] = pos[3 * j + 2];
        jq[tid] = wQ[j] - s_mean[bj];
        js[tid] = wS[j];
        jr[tid] = wR[j];
        jb[tid] = bj;
        // speculative prefetch of next tile (emits global_prefetch_b8)
        if (jt + 1 < N_ATOMS / TILE_J)
            __builtin_prefetch(&pos[3 * (j + TILE_J)], 0, 1);
        __syncthreads();

        #pragma unroll 4
        for (int t = 0; t < TILE_J; ++t) {
            const int jg = jt * TILE_J + t;
            if (jb[t] == bi && jg != i) {
                const float dx = xi - jx[t];
                const float dy = yi - jy[t];
                const float dz = zi - jz[t];
                const float d2 = dx * dx + dy * dy + dz * dz;
                const float dist = sqrtf(d2 + 1e-8f);
                const float ee = KE_C * qi * jq[t] * fast_rcp(dist + 1e-8f)
                               * erff(dist * inv_s);
                const float rr = ri * jr[t];
                const float d6 = d2 * d2 * d2;
                const float ev = -(si * js[t]) * fast_rcp(d6 + rr * rr * rr + 1e-8f);
                acc += 0.5f * (ee + ev);
            }
        }
    }

    red[tid] = acc;
    __syncthreads();
    for (int s = 128; s > 0; s >>= 1) {
        if (tid < s) red[tid] += red[tid + s];
        __syncthreads();
    }
    if (tid == 0) atomicAdd(out, red[0]);
}

// ---------------------------------------------------------------------------
// launcher
// ---------------------------------------------------------------------------
extern "C" void kernel_launch(void* const* d_in, const int* in_sizes, int n_in,
                              void* d_out, int out_size, void* d_ws, size_t ws_size,
                              hipStream_t stream) {
    (void)in_sizes; (void)n_in; (void)out_size; (void)ws_size;

    const float* h0    = (const float*)d_in[0];
    // d_in[1] = h1 (N,3,F) : unused by the reference
    const float* pos   = (const float*)d_in[2];
    const int*   batch = (const int*)  d_in[3];
    const float* qW1   = (const float*)d_in[4];
    const float* qb1   = (const float*)d_in[5];
    const float* qW2   = (const float*)d_in[6];
    const float* vW1   = (const float*)d_in[7];
    const float* vb1   = (const float*)d_in[8];
    const float* vW2   = (const float*)d_in[9];
    const float* vb2   = (const float*)d_in[10];
    const float* sigma = (const float*)d_in[11];

    float* out = (float*)d_out;

    float* ws   = (float*)d_ws;
    float* wQ   = ws;                       // N raw charges
    float* wS   = ws + N_ATOMS;             // N sqrt(c6)
    float* wR   = ws + 2 * N_ATOMS;         // N rv
    float* wQs  = ws + 3 * N_ATOMS;         // G graph q-sums
    float* wCnt = ws + 3 * N_ATOMS + NGRAPH;// G graph counts

    llr_init_kernel<<<1, 64, 0, stream>>>(out, wQs, wCnt);

    llr_head_kernel<<<N_ATOMS / 16, 256, 0, stream>>>(
        h0, batch, qW1, qb1, qW2, vW1, vb1, vW2, vb2,
        wQ, wS, wR, wQs, wCnt);

    llr_pair_kernel<<<N_ATOMS / 256, 256, 0, stream>>>(
        pos, batch, wQ, wS, wR, wQs, wCnt, sigma, out);
}
